// DPF_26207890440348
// MI455X (gfx1250) — matrix operand order repaired and checked by reading, not compiled, not run on hardware
//
#include <hip/hip_runtime.h>
#include <hip/hip_bf16.h>
#include <math.h>

// ---------------------------------------------------------------------------
// CDNA5 / gfx1250 differentiable particle filter
//   encoder: 3x (im2col conv -> WMMA f16 GEMM) -> FC WMMA GEMM with
//            async-LDS double-buffered A panels
//   scan:    per-step ESS kernel (batch-coupled resample decision) +
//            per-batch step kernel with in-LDS WMMA particle encoder
// ---------------------------------------------------------------------------

typedef __attribute__((ext_vector_type(16))) _Float16 v16h;
typedef __attribute__((ext_vector_type(8)))  _Float16 h8;
typedef __attribute__((ext_vector_type(8)))  float    v8f;
typedef int i4 __attribute__((vector_size(16)));   // matches async-LDS builtin param

#define DEV __device__ __forceinline__

#if __has_builtin(__builtin_amdgcn_global_load_async_to_lds_b128) && \
    __has_builtin(__builtin_amdgcn_s_wait_asynccnt)
#define HAVE_ASYNC_LDS 1
#endif

// WMMA wrapper: D = A(16x32 f16) * B(32x16 f16) + C(16x16 f32)
DEV v8f wmma_f16(v16h a, v16h b, v8f c) {
  return __builtin_amdgcn_wmma_f32_16x16x32_f16(false, a, false, b, (short)0, c, false, false);
}

// A-matrix half index -> K offset within a 32-wide K chunk.
// lanes 0-15 : halves 0-7 -> K 0-7,  halves 8-15 -> K 16-23
// lanes 16-31: halves 0-7 -> K 8-15, halves 8-15 -> K 24-31
DEV int a_klocal(int j, bool hi) { return ((j < 8) ? j : j + 8) + (hi ? 8 : 0); }

#ifdef HAVE_ASYNC_LDS
// 16B global -> LDS async copy (per-lane), tracked by ASYNCcnt
DEV void async_copy16(const void* g, void* l) {
  __builtin_amdgcn_global_load_async_to_lds_b128(
      (__attribute__((address_space(1))) i4*)(i4*)(size_t)g,
      (__attribute__((address_space(3))) i4*)(i4*)l, 0, 0);
}
#endif

// ------------------------------- RNG ---------------------------------------
DEV unsigned ihash(unsigned x) {
  x ^= x >> 16; x *= 0x7feb352du;
  x ^= x >> 15; x *= 0x846ca68bu;
  x ^= x >> 16; return x;
}
DEV float u01(unsigned seed, unsigned a, unsigned b, unsigned c) {
  unsigned h = ihash(seed ^ (0x9e3779b9u * (a + 1u)));
  h = ihash(h + 0x85ebca6bu * (b + 1u));
  h = ihash(h + 0xc2b2ae35u * (c + 1u));
  return (float)(h >> 8) * (1.0f / 16777216.0f);
}
DEV float gaussian(unsigned seed, unsigned a, unsigned b, unsigned c) {
  float u1 = fmaxf(u01(seed, a, b, 2u * c), 1e-7f);
  float u2 = u01(seed, a, b, 2u * c + 1u);
  return sqrtf(-2.0f * logf(u1)) * cosf(6.2831853f * u2);
}

// --------------------- weight packing into WMMA-B layout --------------------
// dst[((kc*ntiles + nt)*32 + lane)*16 + j]:
//   lane<16 : col = nt*16+lane,    K = kc*32 + (j<8 ? j : j+8)
//   lane>=16: col = nt*16+lane-16, K = kc*32 + 8 + (j<8 ? j : j+8)
__global__ void pack_conv_w(const float* __restrict__ w, _Float16* __restrict__ dst,
                            int Cin, int Cout, int nchunks, int ntiles) {
  int idx = blockIdx.x * blockDim.x + threadIdx.x;
  int total = nchunks * ntiles * 32 * 16;
  if (idx >= total) return;
  int j = idx & 15;
  int lane = (idx >> 4) & 31;
  int nt = (idx >> 9) % ntiles;
  int kc = idx / (512 * ntiles);
  int col = nt * 16 + (lane & 15);
  int k = kc * 32 + ((j < 8) ? j : j + 8) + ((lane < 16) ? 0 : 8);
  int Kreal = 9 * Cin;
  float v = 0.0f;
  if (k < Kreal && col < Cout) {
    int c = k % Cin, kyx = k / Cin;
    int ky = kyx / 3, kx = kyx % 3;
    v = w[(size_t)(col * Cin + c) * 9 + ky * 3 + kx];   // OIHW
  }
  dst[idx] = (_Float16)v;
}

// FC weight: rows are NCHW-flat (c*256 + y*16 + x); our activations are NHWC,
// so permute rows: k_nhwc = (y*16 + x)*64 + c  ->  orig = c*256 + y*16 + x
__global__ void pack_fc_w(const float* __restrict__ fw, _Float16* __restrict__ dst) {
  int idx = blockIdx.x * blockDim.x + threadIdx.x;   // 512 chunks * 8 ntiles * 512
  if (idx >= 512 * 8 * 512) return;
  int j = idx & 15;
  int lane = (idx >> 4) & 31;
  int nt = (idx >> 9) & 7;
  int kc = idx >> 12;
  int col = nt * 16 + (lane & 15);
  int k = kc * 32 + ((j < 8) ? j : j + 8) + ((lane < 16) ? 0 : 8);       // NHWC-flat, < 16384
  int c = k & 63, x = (k >> 6) & 15, y = k >> 10;
  int orig = c * 256 + y * 16 + x;
  dst[idx] = (_Float16)fw[(size_t)orig * 128 + col];
}

__global__ void pack_pw2(const float* __restrict__ pw2, _Float16* __restrict__ dst) {
  int idx = blockIdx.x * blockDim.x + threadIdx.x;   // 4 chunks * 8 ntiles * 512
  if (idx >= 4 * 8 * 512) return;
  int j = idx & 15;
  int lane = (idx >> 4) & 31;
  int kc = idx >> 12;
  int col = ((idx >> 9) & 7) * 16 + (lane & 15);
  int k = kc * 32 + ((j < 8) ? j : j + 8) + ((lane < 16) ? 0 : 8);
  dst[idx] = (_Float16)pw2[(size_t)k * 128 + col];
}

// ------------------------------ conv1 (f32 in) ------------------------------
// (1600,128,128,3) f32 -> (1600,64,64,16) f16 NHWC, SAME stride-2 3x3
__global__ void conv1_kernel(const float* __restrict__ img, const _Float16* __restrict__ wp,
                             const float* __restrict__ bias, _Float16* __restrict__ out) {
  int bt = blockIdx.y;
  int wave = threadIdx.x >> 5, lane = threadIdx.x & 31;
  int chunk = blockIdx.x * 4 + wave;              // 256 chunks of 16 pixels
  int m = lane & 15; bool hi = lane >= 16;
  int pix = chunk * 16 + m;
  int oy = pix >> 6, ox = pix & 63;
  const float* src = img + (size_t)bt * 128 * 128 * 3;
  v16h a;
#pragma unroll
  for (int j = 0; j < 16; ++j) {
    int k = a_klocal(j, hi);
    float v = 0.0f;
    if (k < 27) {
      int c = k % 3, kyx = k / 3;
      int iy = 2 * oy + kyx / 3, ix = 2 * ox + kyx % 3;
      if (iy < 128 && ix < 128) v = src[(size_t)(iy * 128 + ix) * 3 + c];
    }
    a[j] = (_Float16)v;
  }
  v16h b = *(const v16h*)(wp + (size_t)lane * 16);
  v8f c = {};
  c = wmma_f16(a, b, c);
  int col = lane & 15;
  _Float16* dst = out + (size_t)bt * 64 * 64 * 16;
  float bv = bias[col];
#pragma unroll
  for (int r = 0; r < 8; ++r) {
    int mr = r + (hi ? 8 : 0);
    float v = c[r] + bv;
    dst[(size_t)(chunk * 16 + mr) * 16 + col] = (_Float16)(v > 0.0f ? v : 0.0f);
  }
}

// -------------------------- conv2/conv3 (f16 in) ----------------------------
template <int NT, int CIN, int NCH>
__global__ void conv_f16_kernel(const _Float16* __restrict__ in, const _Float16* __restrict__ wp,
                                const float* __restrict__ bias, _Float16* __restrict__ out,
                                int Hin, int Hout) {
  int bt = blockIdx.y;
  int wave = threadIdx.x >> 5, lane = threadIdx.x & 31;
  int chunk = blockIdx.x * 4 + wave;
  int m = lane & 15; bool hi = lane >= 16;
  int pix = chunk * 16 + m;
  int oy = pix / Hout, ox = pix % Hout;
  const int COUT = NT * 16;
  const int Kreal = 9 * CIN;
  const _Float16* src = in + (size_t)bt * Hin * Hin * CIN;
  v8f acc[NT] = {};
  for (int kc = 0; kc < NCH; ++kc) {
    v16h a;
#pragma unroll
    for (int j = 0; j < 16; ++j) {
      int k = kc * 32 + a_klocal(j, hi);
      _Float16 v = (_Float16)0.0f;
      if (k < Kreal) {
        int c = k % CIN, kyx = k / CIN;
        int iy = 2 * oy + kyx / 3, ix = 2 * ox + kyx % 3;
        if (iy < Hin && ix < Hin) v = src[(size_t)(iy * Hin + ix) * CIN + c];
      }
      a[j] = v;
    }
#pragma unroll
    for (int nt = 0; nt < NT; ++nt) {
      v16h b = *(const v16h*)(wp + ((size_t)(kc * NT + nt) * 32 + lane) * 16);
      acc[nt] = wmma_f16(a, b, acc[nt]);
    }
  }
  _Float16* dst = out + (size_t)bt * Hout * Hout * COUT;
#pragma unroll
  for (int nt = 0; nt < NT; ++nt) {
    int col = nt * 16 + (lane & 15);
    float bv = bias[col];
#pragma unroll
    for (int r = 0; r < 8; ++r) {
      int mr = r + (hi ? 8 : 0);
      float v = acc[nt][r] + bv;
      dst[(size_t)(chunk * 16 + mr) * COUT + col] = (_Float16)(v > 0.0f ? v : 0.0f);
    }
  }
}

// ------------------------------ FC: enc = X @ fw + fb ------------------------
// X: (1600, 16384) f16 NHWC-flat; fw prepacked; enc f32 (1600,128)
// One block per 16-row M-tile; A panels (16 x 256) double-buffered in LDS via
// GLOBAL_LOAD_ASYNC_TO_LDS_B128 (ASYNCcnt) when available; each wave owns
// N-tiles {w, w+4}, so every LDS A fragment feeds two WMMAs.
__global__ void __launch_bounds__(128)
fc_kernel(const _Float16* __restrict__ X, const _Float16* __restrict__ wp,
          const float* __restrict__ fb, float* __restrict__ enc) {
  const int NSC = 64;                       // 64 super-chunks of 256 K-halves
  __shared__ __align__(16) _Float16 sA[2][16 * 256];   // 2 x 8 KB

  int mtile = blockIdx.x;                   // 100 M-tiles
  int tid = threadIdx.x;
  int wave = tid >> 5, lane = tid & 31;
  int m = lane & 15; bool hi = lane >= 16;

  // panel staging: 512 x 16B transfers per buffer, 4 per thread
  auto stage = [&](int kb, int buf) {
#pragma unroll
    for (int i = 0; i < 4; ++i) {
      int idx = tid + 128 * i;
      int r = idx >> 5, seg = idx & 31;
      const _Float16* src = X + (size_t)(mtile * 16 + r) * 16384 + kb * 256 + seg * 8;
      _Float16* dst = &sA[buf][r * 256 + seg * 8];
#ifdef HAVE_ASYNC_LDS
      async_copy16(src, dst);
#else
      *(h8*)dst = *(const h8*)src;
#endif
    }
  };

  v8f acc[2] = {};
  stage(0, 0);
  for (int kb = 0; kb < NSC; ++kb) {
    int cur = kb & 1;
    if (kb + 1 < NSC) stage(kb + 1, 1 - cur);
#ifdef HAVE_ASYNC_LDS
    if (kb + 1 < NSC) __builtin_amdgcn_s_wait_asynccnt(4);   // current panel done
    else              __builtin_amdgcn_s_wait_asynccnt(0);
#endif
    __syncthreads();                        // panel visible to all waves
#pragma unroll
    for (int kc2 = 0; kc2 < 8; ++kc2) {
      const _Float16* p = &sA[cur][m * 256 + kc2 * 32 + (hi ? 8 : 0)];
      v16h a;
#pragma unroll
      for (int j = 0; j < 8; ++j) { a[j] = p[j]; a[j + 8] = p[j + 16]; }
      int kcg = kb * 8 + kc2;
#pragma unroll
      for (int s = 0; s < 2; ++s) {
        int nt = wave + 4 * s;
        const _Float16* bp = wp + ((size_t)(kcg * 8 + nt) * 32 + lane) * 16;
        __builtin_prefetch(bp + 8 * 512, 0, 0);   // next chunk's B panel
        v16h bm = *(const v16h*)bp;
        acc[s] = wmma_f16(a, bm, acc[s]);
      }
    }
    __syncthreads();                        // all waves done before overwrite
  }
#pragma unroll
  for (int s = 0; s < 2; ++s) {
    int nt = wave + 4 * s;
    int col = nt * 16 + (lane & 15);
    float bv = fb[col];
#pragma unroll
    for (int r = 0; r < 8; ++r) {
      int mr = r + (hi ? 8 : 0);
      enc[(size_t)(mtile * 16 + mr) * 128 + col] = acc[s][r] + bv;
    }
  }
}

// ------------------------------ particle filter -----------------------------
__global__ void pf_init(const float* __restrict__ ss, float* __restrict__ particles,
                        float* __restrict__ probs, float* __restrict__ vel) {
  int tid = blockIdx.x * blockDim.x + threadIdx.x;
  if (tid < 3200) {
    int b = tid / 100;
    particles[tid * 2 + 0] = ss[b * 4 + 0];
    particles[tid * 2 + 1] = ss[b * 4 + 1];
    probs[tid] = 0.01f;
  }
  if (tid < 32) {
    vel[tid * 2 + 0] = ss[tid * 4 + 2];
    vel[tid * 2 + 1] = ss[tid * 4 + 3];
  }
}

// batch-coupled resample decision: mean_b(1/sum_n w^2) < 0.5*N
__global__ void pf_ess(const float* __restrict__ probs, int* __restrict__ flag) {
  __shared__ float s[32];
  int tid = threadIdx.x;
  if (tid < 32) {
    float acc = 0.0f;
    for (int n = 0; n < 100; ++n) { float w = probs[tid * 100 + n]; acc += w * w; }
    s[tid] = 1.0f / acc;
  }
  __syncthreads();
  if (tid == 0) {
    float mth = 0.0f;
    for (int i = 0; i < 32; ++i) mth += s[i];
    *flag = (mth * (1.0f / 32.0f) < 50.0f) ? 1 : 0;
  }
}

__global__ void __launch_bounds__(128)
pf_step(const float* __restrict__ enc, const float* __restrict__ vel_in,
        const float* __restrict__ pw1, const float* __restrict__ pb1,
        const _Float16* __restrict__ pw2p, const float* __restrict__ pb2,
        float* __restrict__ particles, float* __restrict__ probs, float* __restrict__ vel,
        const int* __restrict__ flag, float* __restrict__ out, int t) {
  int b = blockIdx.x, tid = threadIdx.x;
  __shared__ float sPx[112], sPy[112];
  __shared__ float sW[100], sQ[100], sCum[100], sLogw[100];
  __shared__ float sEnc[128], sTmp[4];
  __shared__ __align__(16) _Float16 sE1[112 * 128];   // relu(p @ pw1 + pb1), f16
  __shared__ __align__(16) _Float16 sEp[112 * 128];   // e1 @ pw2 + pb2, f16

  if (tid < 128) sEnc[tid] = enc[((size_t)b * 50 + t) * 128 + tid];
  if (tid < 100) {
    sW[tid] = probs[b * 100 + tid];
    sPx[tid] = particles[(b * 100 + tid) * 2 + 0];
    sPy[tid] = particles[(b * 100 + tid) * 2 + 1];
  } else if (tid < 112) { sPx[tid] = 0.0f; sPy[tid] = 0.0f; }
  __syncthreads();

  bool dores = (*flag) != 0;
  if (dores) {
    if (tid == 0) {
      float c = 0.0f;
      for (int n = 0; n < 100; ++n) { sQ[n] = 0.5f * sW[n] + 0.005f; c += sQ[n]; sCum[n] = c; }
    }
    __syncthreads();
    float px = 0.0f, py = 0.0f, wn = 0.0f;
    if (tid < 100) {
      float u = u01(7u, (unsigned)t, (unsigned)b, (unsigned)tid) * sCum[99];
      int lo = 0, hi = 100;                   // searchsorted-left on ascending cum
      while (lo < hi) { int mid = (lo + hi) >> 1; if (sCum[mid] < u) lo = mid + 1; else hi = mid; }
      int idx = lo < 99 ? lo : 99;
      px = sPx[idx]; py = sPy[idx]; wn = sW[idx] / sQ[idx];
    }
    __syncthreads();
    if (tid < 100) { sPx[tid] = px; sPy[tid] = py; sQ[tid] = wn; }
    __syncthreads();
    if (tid == 0) { float s = 0.0f; for (int n = 0; n < 100; ++n) s += sQ[n]; sTmp[0] = s; }
    __syncthreads();
    if (tid < 100) sLogw[tid] = logf(sQ[tid] / sTmp[0] + 1e-12f);
  } else {
    if (tid < 100) sLogw[tid] = logf(sW[tid]);
  }
  __syncthreads();

  // propagate: p += vel + 0.1*N(0,1); prior_log - propose_log cancels in logw
  if (tid < 100) {
    float n0 = 0.1f * gaussian(8u, (unsigned)t, (unsigned)b, 2u * tid);
    float n1 = 0.1f * gaussian(8u, (unsigned)t, (unsigned)b, 2u * tid + 1u);
    sPx[tid] += vel[b * 2 + 0] + n0;
    sPy[tid] += vel[b * 2 + 1] + n1;
  }
  __syncthreads();

  // layer1 (K=2 -> plain VALU): e1[n][h] = relu(px*pw1[0,h] + py*pw1[1,h] + pb1[h])
  {
    int h = tid;  // blockDim == 128
    float w0 = pw1[h], w1 = pw1[128 + h], bb = pb1[h];
    for (int n = 0; n < 112; ++n) {
      float e = sPx[n] * w0 + sPy[n] * w1 + bb;
      sE1[n * 128 + h] = (_Float16)(e > 0.0f ? e : 0.0f);
    }
  }
  __syncthreads();

  // layer2 via WMMA: e_p(112x128) = e1(112x128) @ pw2(128x128) + pb2
  {
    int wave = tid >> 5, lane = tid & 31;
    int m = lane & 15; bool hi = lane >= 16;
    for (int mtile = wave; mtile < 7; mtile += 4) {
      int row = mtile * 16 + m;
      const _Float16* rp = sE1 + row * 128 + (hi ? 8 : 0);
      v16h a[4];
#pragma unroll
      for (int kc = 0; kc < 4; ++kc) {
        const _Float16* p = rp + kc * 32;
#pragma unroll
        for (int j = 0; j < 8; ++j) { a[kc][j] = p[j]; a[kc][j + 8] = p[j + 16]; }
      }
#pragma unroll
      for (int nt = 0; nt < 8; ++nt) {
        v8f acc = {};
#pragma unroll
        for (int kc = 0; kc < 4; ++kc) {
          v16h bm = *(const v16h*)(pw2p + ((size_t)(kc * 8 + nt) * 32 + lane) * 16);
          acc = wmma_f16(a[kc], bm, acc);
        }
        int col = nt * 16 + (lane & 15);
        float bv = pb2[col];
#pragma unroll
        for (int r = 0; r < 8; ++r) {
          int mr = r + (hi ? 8 : 0);
          sEp[(mtile * 16 + mr) * 128 + col] = (_Float16)(acc[r] + bv);
        }
      }
    }
  }
  __syncthreads();

  if (tid == 0) {
    float s = 0.0f;
    for (int h = 0; h < 128; ++h) s += sEnc[h] * sEnc[h];
    sTmp[1] = sqrtf(s);
  }
  __syncthreads();

  // cosine likelihood
  if (tid < 100) {
    float dot = 0.0f, nn = 0.0f;
    for (int h = 0; h < 128; ++h) {
      float e = (float)sEp[tid * 128 + h];
      dot += e * sEnc[h]; nn += e * e;
    }
    float cosv = dot / (sqrtf(nn) * sTmp[1] + 1e-8f);
    sLogw[tid] += logf(0.5f * (cosv + 1.0f) + 1e-8f);
  }
  __syncthreads();

  // softmax + 1e-12
  if (tid == 0) { float mx = -1e30f; for (int n = 0; n < 100; ++n) mx = fmaxf(mx, sLogw[n]); sTmp[2] = mx; }
  __syncthreads();
  if (tid < 100) sW[tid] = expf(sLogw[tid] - sTmp[2]);
  __syncthreads();
  if (tid == 0) { float s = 0.0f; for (int n = 0; n < 100; ++n) s += sW[n]; sTmp[3] = s; }
  __syncthreads();
  if (tid < 100) sW[tid] = sW[tid] / sTmp[3] + 1e-12f;
  __syncthreads();

  if (tid == 0) {
    float pxm = 0.0f, pym = 0.0f;
    for (int n = 0; n < 100; ++n) { pxm += sW[n] * sPx[n]; pym += sW[n] * sPy[n]; }
    out[((size_t)b * 50 + t) * 2 + 0] = pxm;
    out[((size_t)b * 50 + t) * 2 + 1] = pym;
    vel[b * 2 + 0] = vel_in[(b * 50 + t) * 2 + 0];
    vel[b * 2 + 1] = vel_in[(b * 50 + t) * 2 + 1];
  }
  if (tid < 100) {
    probs[b * 100 + tid] = sW[tid];
    particles[(b * 100 + tid) * 2 + 0] = sPx[tid];
    particles[(b * 100 + tid) * 2 + 1] = sPy[tid];
  }
}

// ------------------------------- launcher -----------------------------------
extern "C" void kernel_launch(void* const* d_in, const int* in_sizes, int n_in,
                              void* d_out, int out_size, void* d_ws, size_t ws_size,
                              hipStream_t stream) {
  const float* image = (const float*)d_in[0];
  const float* start_state = (const float*)d_in[1];
  const float* vel_input = (const float*)d_in[2];
  const float* cw1 = (const float*)d_in[3];  const float* cb1 = (const float*)d_in[4];
  const float* cw2 = (const float*)d_in[5];  const float* cb2 = (const float*)d_in[6];
  const float* cw3 = (const float*)d_in[7];  const float* cb3 = (const float*)d_in[8];
  const float* fw  = (const float*)d_in[9];  const float* fb  = (const float*)d_in[10];
  const float* pw1 = (const float*)d_in[11]; const float* pb1 = (const float*)d_in[12];
  const float* pw2 = (const float*)d_in[13]; const float* pb2 = (const float*)d_in[14];
  float* out = (float*)d_out;

  char* ws = (char*)d_ws;
  size_t off = 0;
  auto alloc = [&](size_t bytes) -> char* {
    char* p = ws + off;
    off += (bytes + 255) & ~(size_t)255;
    return p;
  };
  _Float16* conv1o = (_Float16*)alloc((size_t)1600 * 64 * 64 * 16 * 2);  // 210 MB
  _Float16* conv2o = (_Float16*)alloc((size_t)1600 * 32 * 32 * 32 * 2);  // 105 MB
  _Float16* conv3o = conv1o;                                             // overlay (conv1 dead)
  float*    enc    = (float*)alloc((size_t)1600 * 128 * 4);
  _Float16* w1p    = (_Float16*)alloc(1 * 1 * 512 * 2);
  _Float16* w2p    = (_Float16*)alloc(5 * 2 * 512 * 2);
  _Float16* w3p    = (_Float16*)alloc(9 * 4 * 512 * 2);
  _Float16* fwp    = (_Float16*)alloc((size_t)512 * 8 * 512 * 2);        // 4 MB
  _Float16* pw2p   = (_Float16*)alloc(4 * 8 * 512 * 2);
  float* particles = (float*)alloc(32 * 100 * 2 * 4);
  float* probsbuf  = (float*)alloc(32 * 100 * 4);
  float* velbuf    = (float*)alloc(32 * 2 * 4);
  int*   flag      = (int*)alloc(256);

  // 1) pack weights into WMMA-B per-lane layout
  pack_conv_w<<<dim3(2), 256, 0, stream>>>(cw1, w1p, 3, 16, 1, 1);
  pack_conv_w<<<dim3(20), 256, 0, stream>>>(cw2, w2p, 16, 32, 5, 2);
  pack_conv_w<<<dim3(72), 256, 0, stream>>>(cw3, w3p, 32, 64, 9, 4);
  pack_fc_w<<<dim3(8192), 256, 0, stream>>>(fw, fwp);
  pack_pw2<<<dim3(64), 256, 0, stream>>>(pw2, pw2p);

  // 2) encoder
  conv1_kernel<<<dim3(64, 1600), 128, 0, stream>>>(image, w1p, cb1, conv1o);
  conv_f16_kernel<2, 16, 5><<<dim3(16, 1600), 128, 0, stream>>>(conv1o, w2p, cb2, conv2o, 64, 32);
  conv_f16_kernel<4, 32, 9><<<dim3(4, 1600), 128, 0, stream>>>(conv2o, w3p, cb3, conv3o, 32, 16);
  fc_kernel<<<dim3(100), 128, 0, stream>>>(conv3o, fwp, fb, enc);

  // 3) particle filter scan (T sequential steps; resample decision is
  //    batch-coupled -> per-step global ESS kernel)
  pf_init<<<dim3(13), 256, 0, stream>>>(start_state, particles, probsbuf, velbuf);
  for (int t = 0; t < 50; ++t) {
    pf_ess<<<dim3(1), 64, 0, stream>>>(probsbuf, flag);
    pf_step<<<dim3(32), 128, 0, stream>>>(enc, vel_input, pw1, pb1, pw2p, pb2,
                                          particles, probsbuf, velbuf, flag, out, t);
  }
  (void)in_sizes; (void)n_in; (void)out_size; (void)ws_size;
}
